// RBlockPE_91225105367735
// MI455X (gfx1250) — matrix operand order repaired and checked by reading, not compiled, not run on hardware
//
#include <hip/hip_runtime.h>
#include <hip/hip_bf16.h>

// ---------- types ----------
typedef _Float16 h16;
typedef __attribute__((ext_vector_type(16))) _Float16 v16h;
typedef __attribute__((ext_vector_type(8)))  _Float16 v8h;
typedef __attribute__((ext_vector_type(8)))  float    v8f;
typedef __attribute__((ext_vector_type(4)))  float    v4f;

// Problem constants (from reference setup_inputs)
#define NB   16      // batch
#define NC   64      // channels
#define NIC  32      // inner channels
#define NQ   4096    // F*T
#define NK   1024    // Fp*Tp (pooled)
#define PSP  1032    // psi LDS pitch (halfs): 2064B rows -> 16B aligned, banks spread
#define YP   40      // y/W LDS pitch (halfs): 80B rows -> 16B aligned, banks spread

// ---------------------------------------------------------------------------
// K0: collapse w_th into theta weights: avec[c] = sum_ic w_th[ic]*theta_w[ic,c]
// ---------------------------------------------------------------------------
__global__ __launch_bounds__(64) void k0_collapse(const float* __restrict__ theta_w,
                                                  const float* __restrict__ theta_b,
                                                  const float* __restrict__ concat_w,
                                                  float* __restrict__ avec) {
  const int c = threadIdx.x;            // 0..63
  float s = 0.f;
  #pragma unroll
  for (int ic = 0; ic < NIC; ++ic) s = fmaf(concat_w[ic], theta_w[ic * NC + c], s);
  avec[c] = s;
  if (c == 0) {
    float bb = 0.f;
    #pragma unroll
    for (int ic = 0; ic < NIC; ++ic) bb = fmaf(concat_w[ic], theta_b[ic], bb);
    avec[NC] = bb;
  }
}

// ---------------------------------------------------------------------------
// K1a: a[b,n] = sum_c avec[c]*x[b,c,n] + abias + wd0*f/63 + wd1*t/63
// ---------------------------------------------------------------------------
__global__ __launch_bounds__(256) void k1a_avec(const float* __restrict__ x,
                                                const float* __restrict__ avec,
                                                const float* __restrict__ concat_w,
                                                float* __restrict__ a_g) {
  const int gid = blockIdx.x * 256 + threadIdx.x;    // 0..65535
  const int b = gid >> 12, n = gid & (NQ - 1);
  const int f = n >> 6, t = n & 63;
  float acc = avec[NC]
            + concat_w[2 * NIC]     * ((float)f * (1.f / 63.f))
            + concat_w[2 * NIC + 1] * ((float)t * (1.f / 63.f));
  const float* xb = x + (size_t)b * NC * NQ + n;
  #pragma unroll 8
  for (int c = 0; c < NC; ++c) acc = fmaf(avec[c], xb[(size_t)c * NQ], acc);
  a_g[gid] = acc;
}

// ---------------------------------------------------------------------------
// K1b: psi/phi 1x1 conv + 2x2 maxpool; psi -> f16 [b][ic][m]; c[b,m] scalar
// ---------------------------------------------------------------------------
__global__ __launch_bounds__(128) void k1b_pool(const float* __restrict__ x,
                                                const float* __restrict__ psi_w,
                                                const float* __restrict__ psi_b,
                                                const float* __restrict__ phi_w,
                                                const float* __restrict__ phi_b,
                                                const float* __restrict__ concat_w,
                                                h16* __restrict__ psi_g,
                                                float* __restrict__ c_g) {
  __shared__ float sw[2 * NIC * NC + 2 * NIC];       // psi_w | phi_w | psi_b | phi_b
  for (int i = threadIdx.x; i < NIC * NC; i += 128) { sw[i] = psi_w[i]; sw[NIC * NC + i] = phi_w[i]; }
  if (threadIdx.x < NIC) {
    sw[2 * NIC * NC + threadIdx.x]       = psi_b[threadIdx.x];
    sw[2 * NIC * NC + NIC + threadIdx.x] = phi_b[threadIdx.x];
  }
  __syncthreads();

  const int gid = blockIdx.x * 128 + threadIdx.x;    // 0..16383
  const int b = gid >> 10, m = gid & (NK - 1);
  const int fp = m >> 5, tp = m & 31;
  const float* xb = x + (size_t)b * NC * NQ;

  // ---- psi pass ----
  float mx[NIC];
  #pragma unroll
  for (int oc = 0; oc < NIC; ++oc) mx[oc] = -3.0e38f;
  for (int pos = 0; pos < 4; ++pos) {
    const int f = 2 * fp + (pos >> 1), t = 2 * tp + (pos & 1);
    float acc[NIC];
    #pragma unroll
    for (int oc = 0; oc < NIC; ++oc) acc[oc] = sw[2 * NIC * NC + oc];
    for (int c = 0; c < NC; ++c) {
      const float xv = xb[(size_t)c * NQ + f * 64 + t];
      #pragma unroll
      for (int oc = 0; oc < NIC; ++oc) acc[oc] = fmaf(sw[oc * NC + c], xv, acc[oc]);
    }
    #pragma unroll
    for (int oc = 0; oc < NIC; ++oc) mx[oc] = fmaxf(mx[oc], acc[oc]);
  }
  #pragma unroll
  for (int oc = 0; oc < NIC; ++oc)
    psi_g[((size_t)b * NIC + oc) * NK + m] = (h16)mx[oc];

  // ---- phi pass ----
  #pragma unroll
  for (int oc = 0; oc < NIC; ++oc) mx[oc] = -3.0e38f;
  for (int pos = 0; pos < 4; ++pos) {
    const int f = 2 * fp + (pos >> 1), t = 2 * tp + (pos & 1);
    float acc[NIC];
    #pragma unroll
    for (int oc = 0; oc < NIC; ++oc) acc[oc] = sw[2 * NIC * NC + NIC + oc];
    for (int c = 0; c < NC; ++c) {
      const float xv = xb[(size_t)c * NQ + f * 64 + t];
      #pragma unroll
      for (int oc = 0; oc < NIC; ++oc) acc[oc] = fmaf(sw[NIC * NC + oc * NC + c], xv, acc[oc]);
    }
    #pragma unroll
    for (int oc = 0; oc < NIC; ++oc) mx[oc] = fmaxf(mx[oc], acc[oc]);
  }
  float cv = 0.f;
  #pragma unroll
  for (int oc = 0; oc < NIC; ++oc) cv = fmaf(concat_w[NIC + oc], mx[oc], cv);
  cv -= concat_w[2 * NIC]     * ((float)fp * (1.f / 31.f))
      + concat_w[2 * NIC + 1] * ((float)tp * (1.f / 31.f));
  c_g[b * NK + m] = cv;
}

// ---------------------------------------------------------------------------
// K2: main attention matmul via WMMA + W projection via second WMMA stage.
// Block = 128 thr (4 waves); each wave owns 16 rows n. 64 rows / block,
// 64 blocks per batch -> grid 1024.
// ---------------------------------------------------------------------------
__global__ __launch_bounds__(128) void k2_main(const float* __restrict__ a_g,
                                               const float* __restrict__ c_g,
                                               const h16*  __restrict__ psi_g,
                                               const float* __restrict__ Ww,
                                               const float* __restrict__ Wb,
                                               float* __restrict__ wy_g,
                                               float* __restrict__ bs_g) {
  __shared__ __align__(16) float c_s[NK];
  __shared__ __align__(16) h16   psi_s[NIC * PSP];
  __shared__ __align__(16) h16   w2_s[NC * YP];
  __shared__ __align__(16) h16   y_s[4 * 16 * YP];
  __shared__ float red_s[2 * NC];

  const int tid  = threadIdx.x;
  const int lane = tid & 31;
  const int wave = tid >> 5;
  const int b    = blockIdx.x >> 6;
  const int tile = blockIdx.x & 63;
  const int n_base = tile * 64;
  const int row = lane & 15;
  const bool hi = lane >= 16;

  // preload c, psi (f16, pitched), W (f32->f16, [co][ic])
  for (int i = tid; i < NK; i += 128) c_s[i] = c_g[b * NK + i];
  for (int i = tid; i < NIC * NK / 8; i += 128) {
    const int ic = i >> 7, m8 = (i & 127) << 3;
    *reinterpret_cast<v8h*>(&psi_s[ic * PSP + m8]) =
        *reinterpret_cast<const v8h*>(&psi_g[((size_t)b * NIC + ic) * NK + m8]);
  }
  for (int i = tid; i < NC * NIC; i += 128) {
    const int co = i >> 5, ic = i & 31;
    w2_s[co * YP + ic] = (h16)Ww[co * 33 + ic];
  }
  red_s[tid] = 0.f;
  __syncthreads();

  const int n_row = n_base + wave * 16 + row;
  const float aval = a_g[b * NQ + n_row];
  const int koffA = hi ? 8 : 0;    // A elem j<8 -> K=koffA+j ; j>=8 -> K=koffA+16+(j-8)
  const int koffB = hi ? 8 : 0;    // B elem j<8 -> K=koffB+j ; j>=8 -> K=koffB+16+(j-8) ; N = lane&15

  // stage-2 B operands (W^T, loop invariant): co block q covers co=16q..16q+15
  v16h B2[4];
  #pragma unroll
  for (int q = 0; q < 4; ++q) {
    const h16* p = &w2_s[(q * 16 + row) * YP + koffB];
    v8h lo  = *reinterpret_cast<const v8h*>(p);
    v8h hi8 = *reinterpret_cast<const v8h*>(p + 16);
    B2[q] = __builtin_shufflevector(lo, hi8, 0,1,2,3,4,5,6,7,8,9,10,11,12,13,14,15);
  }

  v8f acc0 = {0.f,0.f,0.f,0.f,0.f,0.f,0.f,0.f};
  v8f acc1 = {0.f,0.f,0.f,0.f,0.f,0.f,0.f,0.f};

  for (int k0 = 0; k0 < NK; k0 += 32) {
    // A = relu(a + c) tile, built on the fly (never materialized in HBM)
    const v4f c0 = *reinterpret_cast<const v4f*>(&c_s[k0 + koffA]);
    const v4f c1 = *reinterpret_cast<const v4f*>(&c_s[k0 + koffA + 4]);
    const v4f c2 = *reinterpret_cast<const v4f*>(&c_s[k0 + koffA + 16]);
    const v4f c3 = *reinterpret_cast<const v4f*>(&c_s[k0 + koffA + 20]);
    v16h A;
    #pragma unroll
    for (int j = 0; j < 4; ++j) {
      A[j]      = (h16)fmaxf(aval + c0[j], 0.f);
      A[4 + j]  = (h16)fmaxf(aval + c1[j], 0.f);
      A[8 + j]  = (h16)fmaxf(aval + c2[j], 0.f);
      A[12 + j] = (h16)fmaxf(aval + c3[j], 0.f);
    }
    const h16* pb0 = &psi_s[row * PSP + k0 + koffB];
    const h16* pb1 = &psi_s[(16 + row) * PSP + k0 + koffB];
    v8h b0l = *reinterpret_cast<const v8h*>(pb0);
    v8h b0h = *reinterpret_cast<const v8h*>(pb0 + 16);
    v8h b1l = *reinterpret_cast<const v8h*>(pb1);
    v8h b1h = *reinterpret_cast<const v8h*>(pb1 + 16);
    v16h B0 = __builtin_shufflevector(b0l, b0h, 0,1,2,3,4,5,6,7,8,9,10,11,12,13,14,15);
    v16h B1 = __builtin_shufflevector(b1l, b1h, 0,1,2,3,4,5,6,7,8,9,10,11,12,13,14,15);
    acc0 = __builtin_amdgcn_wmma_f32_16x16x32_f16(false, A, false, B0, (short)0, acc0, false, false);
    acc1 = __builtin_amdgcn_wmma_f32_16x16x32_f16(false, A, false, B1, (short)0, acc1, false, false);
  }

  // stage y tile (scaled by 1/w) to LDS as f16 in A-operand-friendly layout
  h16* ys = &y_s[wave * 16 * YP];
  const int rbase = hi ? 8 : 0;
  #pragma unroll
  for (int j = 0; j < 8; ++j) {
    ys[(rbase + j) * YP + row]      = (h16)(acc0[j] * (1.f / 1024.f));
    ys[(rbase + j) * YP + 16 + row] = (h16)(acc1[j] * (1.f / 1024.f));
  }
  // per-wave private region: same-wave LDS ordering suffices (no barrier)
  const h16* pa2 = &ys[row * YP + koffA];
  v8h a2l = *reinterpret_cast<const v8h*>(pa2);
  v8h a2h = *reinterpret_cast<const v8h*>(pa2 + 16);
  v16h A2 = __builtin_shufflevector(a2l, a2h, 0,1,2,3,4,5,6,7,8,9,10,11,12,13,14,15);

  v8f wacc[4];
  #pragma unroll
  for (int q = 0; q < 4; ++q) {
    v8f z = {0.f,0.f,0.f,0.f,0.f,0.f,0.f,0.f};
    wacc[q] = __builtin_amdgcn_wmma_f32_16x16x32_f16(false, A2, false, B2[q], (short)0, z, false, false);
  }

  // epilogue: +bias, store wy, per-channel BN partial sums
  const int n0 = n_base + wave * 16 + rbase;
  #pragma unroll
  for (int q = 0; q < 4; ++q) {
    const int co = q * 16 + row;
    const float bias = Wb[co];
    float s = 0.f, s2 = 0.f;
    v4f st0, st1;
    #pragma unroll
    for (int j = 0; j < 8; ++j) {
      const float v = wacc[q][j] + bias;
      s += v; s2 = fmaf(v, v, s2);
      if (j < 4) st0[j] = v; else st1[j - 4] = v;
    }
    float* dst = &wy_g[((size_t)b * NC + co) * NQ + n0];
    *reinterpret_cast<v4f*>(dst)     = st0;
    *reinterpret_cast<v4f*>(dst + 4) = st1;
    atomicAdd(&red_s[co], s);
    atomicAdd(&red_s[NC + co], s2);
  }
  __syncthreads();
  bs_g[(size_t)blockIdx.x * 2 * NC + tid] = red_s[tid];
}

// ---------------------------------------------------------------------------
// K3: reduce per-block BN partials -> scale/shift per channel
// ---------------------------------------------------------------------------
__global__ __launch_bounds__(128) void k3_stats(const float* __restrict__ bs_g,
                                                const float* __restrict__ gamma,
                                                const float* __restrict__ beta,
                                                float* __restrict__ stats) {
  __shared__ float tot[2 * NC];
  float s = 0.f;
  for (int blk = 0; blk < 1024; ++blk) s += bs_g[(size_t)blk * 2 * NC + threadIdx.x];
  tot[threadIdx.x] = s;
  __syncthreads();
  if (threadIdx.x < NC) {
    const float inv_n = 1.f / (float)(NB * NQ);
    const float mean = tot[threadIdx.x] * inv_n;
    const float var  = tot[NC + threadIdx.x] * inv_n - mean * mean;
    const float scale = gamma[threadIdx.x] * rsqrtf(var + 1e-5f);
    stats[threadIdx.x]      = scale;
    stats[NC + threadIdx.x] = beta[threadIdx.x] - mean * scale;
  }
}

// ---------------------------------------------------------------------------
// K4: out = wy*scale + shift + x   (vec4)
// ---------------------------------------------------------------------------
__global__ __launch_bounds__(256) void k4_final(const float* __restrict__ wy,
                                                const float* __restrict__ stats,
                                                const float* __restrict__ x,
                                                float* __restrict__ out) {
  const int i4 = blockIdx.x * 256 + threadIdx.x;     // 0..1048575
  const int c = (i4 >> 10) & (NC - 1);
  const float scale = stats[c], shift = stats[NC + c];
  const v4f w  = *reinterpret_cast<const v4f*>(wy + (size_t)i4 * 4);
  const v4f xv = *reinterpret_cast<const v4f*>(x  + (size_t)i4 * 4);
  v4f o;
  #pragma unroll
  for (int j = 0; j < 4; ++j) o[j] = fmaf(w[j], scale, shift) + xv[j];
  *reinterpret_cast<v4f*>(out + (size_t)i4 * 4) = o;
}

// ---------------------------------------------------------------------------
extern "C" void kernel_launch(void* const* d_in, const int* in_sizes, int n_in,
                              void* d_out, int out_size, void* d_ws, size_t ws_size,
                              hipStream_t stream) {
  const float* x        = (const float*)d_in[0];
  const float* psi_w    = (const float*)d_in[1];
  const float* psi_b    = (const float*)d_in[2];
  const float* theta_w  = (const float*)d_in[3];
  const float* theta_b  = (const float*)d_in[4];
  const float* phi_w    = (const float*)d_in[5];
  const float* phi_b    = (const float*)d_in[6];
  const float* concat_w = (const float*)d_in[7];
  const float* W_w      = (const float*)d_in[8];
  const float* W_b      = (const float*)d_in[9];
  const float* bn_gamma = (const float*)d_in[10];
  const float* bn_beta  = (const float*)d_in[11];
  float* out = (float*)d_out;

  char* ws = (char*)d_ws;
  float* avec  = (float*)(ws);                       //   65 f32 (pad 512B)
  float* a_g   = (float*)(ws + 512);                 //   65536 f32
  float* c_g   = (float*)(ws + 512 + 262144);        //   16384 f32
  h16*   psi_g = (h16*)  (ws + 328192);              //   16*32*1024 f16 = 1 MB
  float* wy_g  = (float*)(ws + 1376768);             //   16.78 MB
  float* bs_g  = (float*)(ws + 18153984);            //   1024*128 f32
  float* stats = (float*)(ws + 18678272);            //   128 f32

  k0_collapse<<<1, 64, 0, stream>>>(theta_w, theta_b, concat_w, avec);
  k1a_avec<<<256, 256, 0, stream>>>(x, avec, concat_w, a_g);
  k1b_pool<<<128, 128, 0, stream>>>(x, psi_w, psi_b, phi_w, phi_b, concat_w, psi_g, c_g);
  k2_main<<<1024, 128, 0, stream>>>(a_g, c_g, psi_g, W_w, W_b, wy_g, bs_g);
  k3_stats<<<1, 128, 0, stream>>>(bs_g, bn_gamma, bn_beta, stats);
  k4_final<<<4096, 256, 0, stream>>>(wy_g, stats, x, out);
}
